// ProteinLigandTNP_34746285425455
// MI455X (gfx1250) — matrix operand order repaired and checked by reading, not compiled, not run on hardware
//
#include <hip/hip_runtime.h>
#include <math.h>

// ---------------------------------------------------------------------------
// MI455X / gfx1250 implementation of the ProteinLigand TNP reference.
// All GEMMs use v_wmma_f32_16x16x32_bf16 (bf16 A/B, f32 accumulate).
// Attention is flash-style (no materialized 2048x2048 logits).
// A-fragment loads are 128-bit LDS vector loads (ds_load_b128), exploiting the
// ISA fragment layouts: A-frag = two contiguous 8x bf16 runs per lane,
// B-frag = 16 strided u16 LDS reads per lane, same K order as A.
//
// Input flattening assumption (JAX pytree leaf order: top-level dict in
// insertion order for the 7 arrays, then 'params' flattened with dict keys
// sorted alphabetically, lists in index order):
//   0 ctx_protein  1 ctx_drug  2 ctx_affinity  3 qry_protein  4 qry_drug
//   5 ctx_ppr  6 ctx_trust
//   7 cold_start_bias
//   8 ctx_affinity_proj.b   9 ctx_affinity_proj.w
//  10 ctx_drug_proj.b      11 ctx_drug_proj.w
//  12 ctx_fusion.b         13 ctx_fusion.w
//  14 ctx_protein_proj.b   15 ctx_protein_proj.w
//  16 density_proj.b       17 density_proj.w
//  18 final_norm.b         19 final_norm.g
//  20 head1.b              21 head1.w
//  22 head2.b              23 head2.w
//  24 + 14*l + {0 ff1.b, 1 ff1.w, 2 ff2.b, 3 ff2.w, 4 in_proj.b, 5 in_proj.w,
//               6 log_ppr_alpha, 7 norm1.b, 8 norm1.g, 9 norm2.b, 10 norm2.g,
//               11 out_proj.b, 12 out_proj.w, 13 trust_scale}   (l = 0..3)
//  80 qry_drug_proj.b  81 qry_drug_proj.w  82 qry_fusion.b  83 qry_fusion.w
//  84 qry_protein_proj.b  85 qry_protein_proj.w  86 type_embed
// ---------------------------------------------------------------------------

typedef __bf16 bf16_t;
typedef __attribute__((ext_vector_type(16))) __bf16 v16bf;
typedef __attribute__((ext_vector_type(8)))  float  v8f;

#define WMMA_BF16(a, b, c) \
  __builtin_amdgcn_wmma_f32_16x16x32_bf16(false, (a), false, (b), (short)0, (c), false, false)

union frag16u {
  v16bf v;
  uint4 q[2];
};

__device__ __forceinline__ v8f v8f_zero() {
  v8f z;
#pragma unroll
  for (int i = 0; i < 8; ++i) z[i] = 0.0f;
  return z;
}

// A fragment: 16x32 bf16 (M = lane&15). Per the ISA table, each lane's 16
// elements are two contiguous 8-element K runs at K = half*8 and 16 + half*8.
// Both runs are 16B-aligned for our strides -> two ds_load_b128.
__device__ __forceinline__ v16bf ldsA_frag(const bf16_t* base, int ld, int lane) {
  const int half = lane >> 4, m = lane & 15;
  frag16u f;
  f.q[0] = *(const uint4*)&base[m * ld + half * 8];
  f.q[1] = *(const uint4*)&base[m * ld + 16 + half * 8];
  return f.v;
}
// B fragment: 32x16 bf16, N = lane&15, elements = K in the same order as the
// A fragment (K = half*8 + s, 16 + half*8 + s); row-major (K x N) LDS.
__device__ __forceinline__ v16bf ldsB_frag(const bf16_t* base, int ld, int lane) {
  const int half = lane >> 4, n = lane & 15;
  v16bf v;
#pragma unroll
  for (int s = 0; s < 8; ++s) {
    v[s] = base[(half * 8 + s) * ld + n];
    v[8 + s] = base[(16 + half * 8 + s) * ld + n];
  }
  return v;
}

// ---------------------------------------------------------------------------
// Generic bf16 GEMM:  out = act(A[MxK] @ W[KxN] + bias + extra) (+ resid)
// Block = 128 threads (4 waves), 64x64 output tile, k-step 32.
// ACT: 0 none, 1 exact GELU, 2 ReLU.
// ---------------------------------------------------------------------------
template <int ACT>
__global__ __launch_bounds__(128) void gemm_bf16_k(
    const bf16_t* __restrict__ A, const bf16_t* __restrict__ W,
    const float* __restrict__ bias, const float* __restrict__ extra,
    const float* __restrict__ resid, int ldr,
    float* __restrict__ outF, bf16_t* __restrict__ outB, int ldo,
    int M, int N, int K) {
  __shared__ bf16_t As[64 * 40];
  __shared__ bf16_t Ws[32 * 72];
  const int n0 = blockIdx.x * 64, m0 = blockIdx.y * 64;
  const int t = threadIdx.x, wave = t >> 5, lane = t & 31;
  (void)M;
  v8f acc[4];
#pragma unroll
  for (int i = 0; i < 4; ++i) acc[i] = v8f_zero();

  for (int k0 = 0; k0 < K; k0 += 32) {
#pragma unroll
    for (int c = t; c < 256; c += 128) {  // A tile 64x32
      const int row = c >> 2, cc = (c & 3) * 8;
      *(uint4*)&As[row * 40 + cc] = *(const uint4*)&A[(size_t)(m0 + row) * K + k0 + cc];
    }
#pragma unroll
    for (int c = t; c < 256; c += 128) {  // W tile 32x64
      const int row = c >> 3, cc = (c & 7) * 8;
      *(uint4*)&Ws[row * 72 + cc] = *(const uint4*)&W[(size_t)(k0 + row) * N + n0 + cc];
    }
    __syncthreads();
    const v16bf a = ldsA_frag(&As[wave * 16 * 40], 40, lane);
#pragma unroll
    for (int tt = 0; tt < 4; ++tt) {
      const v16bf b = ldsB_frag(&Ws[tt * 16], 72, lane);
      acc[tt] = WMMA_BF16(a, b, acc[tt]);
    }
    __syncthreads();
  }

  const int half = lane >> 4, cl = lane & 15;
#pragma unroll
  for (int tt = 0; tt < 4; ++tt)
#pragma unroll
    for (int r = 0; r < 8; ++r) {
      const int row = m0 + wave * 16 + r + 8 * half;
      const int col = n0 + tt * 16 + cl;
      float v = acc[tt][r] + bias[col];
      if (extra) v += extra[col];
      if (ACT == 1) v = 0.5f * v * (1.0f + erff(v * 0.70710678118654752f));
      if (ACT == 2) v = fmaxf(v, 0.0f);
      if (resid) v += resid[(size_t)row * ldr + col];
      if (outF) outF[(size_t)row * ldo + col] = v;
      if (outB) outB[(size_t)row * ldo + col] = (bf16_t)v;
    }
}

// ---------------------------------------------------------------------------
// Flash attention. Grid (32 q-tiles, 8 heads), 128 threads (4 waves).
// Q/V layout: [head][2048][64] bf16 (V pre-gated by trust for ctx rows).
// Kt layout:  [head][64][2048] bf16 (pre-transposed so the Q@K^T B-operand is
// a row-major (d x ctx) LDS tile for ldsB_frag, copied in 128-bit chunks).
// Mask structure: every row attends to ctx cols 0..1535 (+ col bias);
// query rows (>=1536) additionally attend to themselves only.
// ---------------------------------------------------------------------------
__global__ __launch_bounds__(128) void flash_attn_k(
    const bf16_t* __restrict__ Qb, const bf16_t* __restrict__ Kb,
    const bf16_t* __restrict__ Kt, const bf16_t* __restrict__ Vb,
    const float* __restrict__ cb, bf16_t* __restrict__ outB) {
  const int qt = blockIdx.x;  // 0..31 (tiles of 64 rows; >=24 are query tiles)
  const int h  = blockIdx.y;  // 0..7
  const int t = threadIdx.x, wave = t >> 5, lane = t & 31;
  const int half = lane >> 4, cl = lane & 15;
  const float scale = 0.125f;  // d=64 -> 1/sqrt(64)
  __shared__ bf16_t Qs[64 * 72];
  __shared__ bf16_t Ks[64 * 72];  // K^T tile: rows = d (0..63), cols = ctx cols
  __shared__ bf16_t Vs[64 * 72];  // rows = ctx cols, cols = d
  __shared__ bf16_t Ps[64 * 72];  // per-wave 16-row slabs
  __shared__ float  cbs[64];
  const size_t hbase = (size_t)h * 2048 * 64;

  {  // load Q tile once
    const bf16_t* Qh = Qb + hbase + (size_t)qt * 64 * 64;
#pragma unroll
    for (int c = t; c < 512; c += 128) {
      const int row = c >> 3, cc = (c & 7) * 8;
      *(uint4*)&Qs[row * 72 + cc] = *(const uint4*)&Qh[row * 64 + cc];
    }
  }
  float mst[8], lst[8];
  v8f o[4];
#pragma unroll
  for (int r = 0; r < 8; ++r) { mst[r] = -INFINITY; lst[r] = 0.0f; }
#pragma unroll
  for (int i = 0; i < 4; ++i) o[i] = v8f_zero();

  for (int kt = 0; kt < 24; ++kt) {  // 1536 ctx cols
    const bf16_t* Vh = Vb + hbase + (size_t)kt * 64 * 64;
#pragma unroll
    for (int c = t; c < 512; c += 128) {
      const int row = c >> 3, cc = (c & 7) * 8;
      // Kt tile: row = d, 64 contiguous ctx cols at this k-tile
      *(uint4*)&Ks[row * 72 + cc] =
          *(const uint4*)&Kt[((size_t)h * 64 + row) * 2048 + kt * 64 + cc];
      *(uint4*)&Vs[row * 72 + cc] = *(const uint4*)&Vh[row * 64 + cc];
    }
    if (t < 64) cbs[t] = cb[kt * 64 + t];
    __syncthreads();

    // S = Q @ K^T  (per-wave 16x64 slab, two 32-deep k-steps over d)
    const v16bf a0 = ldsA_frag(&Qs[wave * 16 * 72], 72, lane);
    const v16bf a1 = ldsA_frag(&Qs[wave * 16 * 72 + 32], 72, lane);
    v8f s[4];
#pragma unroll
    for (int tt = 0; tt < 4; ++tt) {
      v8f z = v8f_zero();
      z = WMMA_BF16(a0, ldsB_frag(&Ks[tt * 16], 72, lane), z);
      z = WMMA_BF16(a1, ldsB_frag(&Ks[32 * 72 + tt * 16], 72, lane), z);
      s[tt] = z;
    }
#pragma unroll
    for (int tt = 0; tt < 4; ++tt) {
      const float cbv = cbs[tt * 16 + cl];
#pragma unroll
      for (int r = 0; r < 8; ++r) s[tt][r] = s[tt][r] * scale + cbv;
    }

    // Online softmax: rows live in 16-lane groups (row = r + 8*half).
#pragma unroll
    for (int r = 0; r < 8; ++r) {
      float x = fmaxf(fmaxf(s[0][r], s[1][r]), fmaxf(s[2][r], s[3][r]));
#pragma unroll
      for (int off = 1; off < 16; off <<= 1) x = fmaxf(x, __shfl_xor(x, off, 32));
      const float mn = fmaxf(mst[r], x);
      const float corr = __expf(mst[r] - mn);
      float psum = 0.0f;
#pragma unroll
      for (int tt = 0; tt < 4; ++tt) {
        const float p = __expf(s[tt][r] - mn);
        psum += p;
        Ps[(wave * 16 + r + 8 * half) * 72 + tt * 16 + cl] = (bf16_t)p;
      }
#pragma unroll
      for (int off = 1; off < 16; off <<= 1) psum += __shfl_xor(psum, off, 32);
      lst[r] = lst[r] * corr + psum;
      mst[r] = mn;
#pragma unroll
      for (int tt = 0; tt < 4; ++tt) o[tt][r] = o[tt][r] * corr;
    }
    asm volatile("s_wait_dscnt 0" ::: "memory");  // Ps stores -> Ps frag loads

    // O += P @ V
    const v16bf pa0 = ldsA_frag(&Ps[wave * 16 * 72], 72, lane);
    const v16bf pa1 = ldsA_frag(&Ps[wave * 16 * 72 + 32], 72, lane);
#pragma unroll
    for (int tt = 0; tt < 4; ++tt) {
      o[tt] = WMMA_BF16(pa0, ldsB_frag(&Vs[tt * 16], 72, lane), o[tt]);
      o[tt] = WMMA_BF16(pa1, ldsB_frag(&Vs[32 * 72 + tt * 16], 72, lane), o[tt]);
    }
    __syncthreads();
  }

  // Self-attention diagonal term for query tiles.
  if (qt >= 24) {
    const bf16_t* Kg = Kb + hbase + (size_t)qt * 64 * 64;
    const bf16_t* Vg = Vb + hbase + (size_t)qt * 64 * 64;
#pragma unroll
    for (int r = 0; r < 8; ++r) {
      const int rowl = wave * 16 + r + 8 * half;
      float dot = 0.0f;
      for (int d = 0; d < 64; ++d)
        dot += (float)Qs[rowl * 72 + d] * (float)Kg[(size_t)rowl * 64 + d];
      const float ss = dot * scale;
      const float mn = fmaxf(mst[r], ss);
      const float corr = __expf(mst[r] - mn);
      const float pe = __expf(ss - mn);
      lst[r] = lst[r] * corr + pe;
      mst[r] = mn;
#pragma unroll
      for (int tt = 0; tt < 4; ++tt)
        o[tt][r] = o[tt][r] * corr + pe * (float)Vg[(size_t)rowl * 64 + tt * 16 + cl];
    }
  }

#pragma unroll
  for (int tt = 0; tt < 4; ++tt)
#pragma unroll
    for (int r = 0; r < 8; ++r) {
      const int grow = qt * 64 + wave * 16 + r + 8 * half;
      const int gcol = h * 64 + tt * 16 + cl;
      outB[(size_t)grow * 512 + gcol] = (bf16_t)(o[tt][r] / lst[r]);
    }
}

// ---------------------------------------------------------------------------
// Small helper kernels
// ---------------------------------------------------------------------------
__global__ void f32_to_bf16_k(const float* __restrict__ src, bf16_t* __restrict__ dst, int n) {
  const int i = blockIdx.x * blockDim.x + threadIdx.x;
  if (i < n) dst[i] = (bf16_t)src[i];
}

// ctx_affinity (1536x1) @ w(1x128) + b -> ctx_feat cols 512..639 (bf16)
__global__ void aff_proj_k(const float* __restrict__ aff, const float* __restrict__ w,
                           const float* __restrict__ b, bf16_t* __restrict__ dst) {
  const int i = blockIdx.x * blockDim.x + threadIdx.x;
  if (i >= 1536 * 128) return;
  const int row = i >> 7, c = i & 127;
  dst[(size_t)row * 640 + 512 + c] = (bf16_t)(aff[row] * w[c] + b[c]);
}

// extra vector for query rows: type_embed[1] + density_proj(1.0)
// density = tanh(1536/32) = tanh(48) == 1.0f in fp32.
__global__ void qextra_k(const float* __restrict__ te, const float* __restrict__ dw,
                         const float* __restrict__ db, float* __restrict__ out) {
  const int j = blockIdx.x * blockDim.x + threadIdx.x;
  if (j < 512) out[j] = te[512 + j] + dw[j] + db[j];
}

__global__ void colbias_k(const float* __restrict__ ppr, const float* __restrict__ alpha,
                          float* __restrict__ cb) {
  const int i = blockIdx.x * blockDim.x + threadIdx.x;
  if (i < 1536) cb[i] = -alpha[0] * logf(fmaxf(ppr[i], 1e-8f));
}

// qkv f32 [2048][1536] -> Q/K/V bf16 [8][2048][64] plus K^T [8][64][2048];
// V gated by trust on ctx rows.
__global__ void repack_qkv_k(const float* __restrict__ qkv, const float* __restrict__ trust,
                             const float* __restrict__ tscale,
                             bf16_t* __restrict__ Qb, bf16_t* __restrict__ Kb,
                             bf16_t* __restrict__ Kt, bf16_t* __restrict__ Vb) {
  const int i = blockIdx.x * blockDim.x + threadIdx.x;
  if (i >= 2048 * 512) return;
  const int row = i >> 9, c = i & 511;
  const int h = c >> 6, d = c & 63;
  const float* qr = qkv + (size_t)row * 1536;
  float gate = 1.0f;
  if (row < 1536) gate = 1.0f / (1.0f + __expf(-tscale[0] * trust[row]));
  const size_t o = ((size_t)h * 2048 + row) * 64 + d;
  const bf16_t kv = (bf16_t)qr[512 + c];
  Qb[o] = (bf16_t)qr[c];
  Kb[o] = kv;
  Kt[((size_t)h * 64 + d) * 2048 + row] = kv;
  Vb[o] = (bf16_t)(qr[1024 + c] * gate);
}

// LayerNorm over 512 cols, one wave32 per row, bf16 output.
__global__ __launch_bounds__(128) void layernorm_k(const float* __restrict__ x,
                                                   const float* __restrict__ g,
                                                   const float* __restrict__ b,
                                                   bf16_t* __restrict__ outB) {
  const int row = blockIdx.x * 4 + (threadIdx.x >> 5);
  const int lane = threadIdx.x & 31;
  const float* xr = x + (size_t)row * 512;
  float s = 0.0f, s2 = 0.0f;
  for (int j = lane; j < 512; j += 32) { const float v = xr[j]; s += v; s2 += v * v; }
#pragma unroll
  for (int off = 1; off < 32; off <<= 1) {
    s += __shfl_xor(s, off, 32);
    s2 += __shfl_xor(s2, off, 32);
  }
  const float mu = s * (1.0f / 512.0f);
  const float var = s2 * (1.0f / 512.0f) - mu * mu;
  const float rs = rsqrtf(var + 1e-5f);
  bf16_t* orow = outB + (size_t)row * 512;
  for (int j = lane; j < 512; j += 32) orow[j] = (bf16_t)((xr[j] - mu) * rs * g[j] + b[j]);
}

__global__ __launch_bounds__(256) void aff_mean_k(const float* __restrict__ aff,
                                                  float* __restrict__ out) {
  __shared__ float red[256];
  float s = 0.0f;
  for (int i = threadIdx.x; i < 1536; i += 256) s += aff[i];
  red[threadIdx.x] = s;
  __syncthreads();
  for (int w = 128; w > 0; w >>= 1) {
    if (threadIdx.x < w) red[threadIdx.x] += red[threadIdx.x + w];
    __syncthreads();
  }
  if (threadIdx.x == 0) out[0] = red[0] * (1.0f / 1536.0f);
}

// head2: per query row, pred = h1row(256) @ w2(256x2) + b2;
// mu = pred0 + mean(ctx_aff); sigma = softplus(pred1) + 1e-4  ((1-density)==0).
__global__ __launch_bounds__(128) void head2_k(const bf16_t* __restrict__ h1,
                                               const float* __restrict__ w2,
                                               const float* __restrict__ b2,
                                               const float* __restrict__ affmean,
                                               float* __restrict__ out) {
  const int row = blockIdx.x * 4 + (threadIdx.x >> 5);
  const int lane = threadIdx.x & 31;
  const bf16_t* hr = h1 + (size_t)row * 256;
  float p0 = 0.0f, p1 = 0.0f;
  for (int j = lane; j < 256; j += 32) {
    const float v = (float)hr[j];
    p0 += v * w2[j * 2 + 0];
    p1 += v * w2[j * 2 + 1];
  }
#pragma unroll
  for (int off = 1; off < 32; off <<= 1) {
    p0 += __shfl_xor(p0, off, 32);
    p1 += __shfl_xor(p1, off, 32);
  }
  if (lane == 0) {
    out[row] = p0 + b2[0] + affmean[0];
    const float ls = p1 + b2[1];
    const float sp = (ls > 20.0f) ? ls : log1pf(__expf(ls));
    out[512 + row] = sp + 1e-4f;
  }
}

// ---------------------------------------------------------------------------
// Host orchestration
// ---------------------------------------------------------------------------
extern "C" void kernel_launch(void* const* d_in, const int* in_sizes, int n_in,
                              void* d_out, int out_size, void* d_ws, size_t ws_size,
                              hipStream_t stream) {
  (void)in_sizes; (void)n_in; (void)out_size; (void)ws_size;

  // layer leaf offsets
  enum { L_FF1_B = 0, L_FF1_W, L_FF2_B, L_FF2_W, L_INP_B, L_INP_W, L_ALPHA,
         L_N1_B, L_N1_G, L_N2_B, L_N2_G, L_OUT_B, L_OUT_W, L_TSCALE };
  auto LP = [&](int l, int off) { return d_in[24 + l * 14 + off]; };

  // workspace allocator (256B aligned)
  char* wp = (char*)d_ws;
  auto alloc = [&](size_t bytes) -> char* {
    char* r = wp;
    wp += (bytes + 255) & ~(size_t)255;
    return r;
  };

  bf16_t* bCtxProt = (bf16_t*)alloc((size_t)1536 * 1280 * 2);
  bf16_t* bCtxDrug = (bf16_t*)alloc((size_t)1536 * 2048 * 2);
  bf16_t* bQryProt = (bf16_t*)alloc((size_t)512 * 1280 * 2);
  bf16_t* bQryDrug = (bf16_t*)alloc((size_t)512 * 2048 * 2);
  bf16_t* wCP = (bf16_t*)alloc((size_t)1280 * 256 * 2);
  bf16_t* wCD = (bf16_t*)alloc((size_t)2048 * 256 * 2);
  bf16_t* wCF = (bf16_t*)alloc((size_t)640 * 512 * 2);
  bf16_t* wQP = (bf16_t*)alloc((size_t)1280 * 256 * 2);
  bf16_t* wQD = (bf16_t*)alloc((size_t)2048 * 256 * 2);
  bf16_t* wQF = (bf16_t*)alloc((size_t)512 * 512 * 2);
  bf16_t* wH1 = (bf16_t*)alloc((size_t)512 * 256 * 2);
  bf16_t *wIN[4], *wOUT[4], *wF1[4], *wF2[4];
  for (int l = 0; l < 4; ++l) {
    wIN[l]  = (bf16_t*)alloc((size_t)512 * 1536 * 2);
    wOUT[l] = (bf16_t*)alloc((size_t)512 * 512 * 2);
    wF1[l]  = (bf16_t*)alloc((size_t)512 * 1024 * 2);
    wF2[l]  = (bf16_t*)alloc((size_t)1024 * 512 * 2);
  }
  bf16_t* ctxFeat = (bf16_t*)alloc((size_t)1536 * 640 * 2);
  bf16_t* qryFeat = (bf16_t*)alloc((size_t)512 * 512 * 2);
  float*  tokens  = (float*)alloc((size_t)2048 * 512 * 4);
  bf16_t* hB      = (bf16_t*)alloc((size_t)2048 * 512 * 2);
  float*  qkv     = (float*)alloc((size_t)2048 * 1536 * 4);
  bf16_t* Qb      = (bf16_t*)alloc((size_t)8 * 2048 * 64 * 2);
  bf16_t* Kb      = (bf16_t*)alloc((size_t)8 * 2048 * 64 * 2);
  bf16_t* KtB     = (bf16_t*)alloc((size_t)8 * 64 * 2048 * 2);
  bf16_t* Vb      = (bf16_t*)alloc((size_t)8 * 2048 * 64 * 2);
  bf16_t* attnB   = (bf16_t*)alloc((size_t)2048 * 512 * 2);
  bf16_t* ffmid   = (bf16_t*)alloc((size_t)2048 * 1024 * 2);
  bf16_t* h1B     = (bf16_t*)alloc((size_t)512 * 256 * 2);
  float*  cbv     = (float*)alloc((size_t)1536 * 4);
  float*  qx      = (float*)alloc((size_t)512 * 4);
  float*  affm    = (float*)alloc(256);

  auto conv = [&](const void* src, bf16_t* dst, int n) {
    f32_to_bf16_k<<<(n + 255) / 256, 256, 0, stream>>>((const float*)src, dst, n);
  };
  auto gemm = [&](int act, const bf16_t* A, const bf16_t* W, const void* bias,
                  const float* extra, const float* resid, int ldr,
                  float* outF, bf16_t* outB, int ldo, int M, int N, int K) {
    dim3 g(N / 64, M / 64), blk(128);
    if (act == 0)
      gemm_bf16_k<0><<<g, blk, 0, stream>>>(A, W, (const float*)bias, extra, resid, ldr,
                                            outF, outB, ldo, M, N, K);
    else if (act == 1)
      gemm_bf16_k<1><<<g, blk, 0, stream>>>(A, W, (const float*)bias, extra, resid, ldr,
                                            outF, outB, ldo, M, N, K);
    else
      gemm_bf16_k<2><<<g, blk, 0, stream>>>(A, W, (const float*)bias, extra, resid, ldr,
                                            outF, outB, ldo, M, N, K);
  };

  // ---- weight / input conversions to bf16 ----
  conv(d_in[0], bCtxProt, 1536 * 1280);
  conv(d_in[1], bCtxDrug, 1536 * 2048);
  conv(d_in[3], bQryProt, 512 * 1280);
  conv(d_in[4], bQryDrug, 512 * 2048);
  conv(d_in[15], wCP, 1280 * 256);
  conv(d_in[11], wCD, 2048 * 256);
  conv(d_in[13], wCF, 640 * 512);
  conv(d_in[85], wQP, 1280 * 256);
  conv(d_in[81], wQD, 2048 * 256);
  conv(d_in[83], wQF, 512 * 512);
  conv(d_in[21], wH1, 512 * 256);
  for (int l = 0; l < 4; ++l) {
    conv(LP(l, L_INP_W), wIN[l], 512 * 1536);
    conv(LP(l, L_OUT_W), wOUT[l], 512 * 512);
    conv(LP(l, L_FF1_W), wF1[l], 512 * 1024);
    conv(LP(l, L_FF2_W), wF2[l], 1024 * 512);
  }

  aff_mean_k<<<1, 256, 0, stream>>>((const float*)d_in[2], affm);
  qextra_k<<<2, 256, 0, stream>>>((const float*)d_in[86], (const float*)d_in[17],
                                  (const float*)d_in[16], qx);

  // ---- input embeddings ----
  gemm(0, bCtxProt, wCP, d_in[14], nullptr, nullptr, 0, nullptr, ctxFeat, 640, 1536, 256, 1280);
  gemm(0, bCtxDrug, wCD, d_in[10], nullptr, nullptr, 0, nullptr, ctxFeat + 256, 640, 1536, 256, 2048);
  aff_proj_k<<<768, 256, 0, stream>>>((const float*)d_in[2], (const float*)d_in[9],
                                      (const float*)d_in[8], ctxFeat);
  gemm(0, bQryProt, wQP, d_in[84], nullptr, nullptr, 0, nullptr, qryFeat, 512, 512, 256, 1280);
  gemm(0, bQryDrug, wQD, d_in[80], nullptr, nullptr, 0, nullptr, qryFeat + 256, 512, 512, 256, 2048);

  // fusion -> tokens (ctx rows get type_embed[0], qry rows get type_embed[1]+density vec)
  gemm(0, ctxFeat, wCF, d_in[12], (const float*)d_in[86], nullptr, 0,
       tokens, nullptr, 512, 1536, 512, 640);
  gemm(0, qryFeat, wQF, d_in[82], qx, nullptr, 0,
       tokens + (size_t)1536 * 512, nullptr, 512, 512, 512, 512);

  // ---- transformer layers ----
  for (int l = 0; l < 4; ++l) {
    layernorm_k<<<512, 128, 0, stream>>>(tokens, (const float*)LP(l, L_N1_G),
                                         (const float*)LP(l, L_N1_B), hB);
    gemm(0, hB, wIN[l], LP(l, L_INP_B), nullptr, nullptr, 0,
         qkv, nullptr, 1536, 2048, 1536, 512);
    colbias_k<<<6, 256, 0, stream>>>((const float*)d_in[5], (const float*)LP(l, L_ALPHA), cbv);
    repack_qkv_k<<<4096, 256, 0, stream>>>(qkv, (const float*)d_in[6],
                                           (const float*)LP(l, L_TSCALE), Qb, Kb, KtB, Vb);
    flash_attn_k<<<dim3(32, 8), 128, 0, stream>>>(Qb, Kb, KtB, Vb, cbv, attnB);
    gemm(0, attnB, wOUT[l], LP(l, L_OUT_B), nullptr, tokens, 512,
         tokens, nullptr, 512, 2048, 512, 512);
    layernorm_k<<<512, 128, 0, stream>>>(tokens, (const float*)LP(l, L_N2_G),
                                         (const float*)LP(l, L_N2_B), hB);
    gemm(1, hB, wF1[l], LP(l, L_FF1_B), nullptr, nullptr, 0,
         nullptr, ffmid, 1024, 2048, 1024, 512);
    gemm(0, ffmid, wF2[l], LP(l, L_FF2_B), nullptr, tokens, 512,
         tokens, nullptr, 512, 2048, 512, 1024);
  }

  // ---- final norm + heads ----
  layernorm_k<<<512, 128, 0, stream>>>(tokens, (const float*)d_in[19],
                                       (const float*)d_in[18], hB);
  gemm(2, hB + (size_t)1536 * 512, wH1, d_in[20], nullptr, nullptr, 0,
       nullptr, h1B, 256, 512, 256, 512);
  head2_k<<<128, 128, 0, stream>>>(h1B, (const float*)d_in[23], (const float*)d_in[22],
                                   affm, (float*)d_out);
}
